// MambaBackbone_75222057222775
// MI455X (gfx1250) — matrix operand order repaired and checked by reading, not compiled, not run on hardware
//
#include <hip/hip_runtime.h>

// ---------------------------------------------------------------------------
// Mamba backbone forward for MI455X (gfx1250, wave32, WMMA).
//   B=8, L=2048, D_MODEL=512, D_INNER=1024, N=16, DT_RANK=32, D_CONV=4, 4 layers
// Strategy:
//   * all GEMMs via v_wmma_f32_16x16x32_bf16 (bf16 in, f32 accumulate)
//   * A tiles staged with GLOBAL_LOAD_ASYNC_TO_LDS_B128, double buffered:
//     next K-tile DMA overlaps current tile's WMMAs; one barrier per K step
//   * weights bf16 in workspace; L2 (192MB) keeps them hot, fragments loaded
//     straight from global
//   * selective scan: lane = (d,n) state, 16-lane shuffle reduction, fused
//     y*silu(z) epilogue; exp via v_exp_f32
// ---------------------------------------------------------------------------

#define B_SZ    8
#define L_SEQ   2048
#define M_TOK   (B_SZ * L_SEQ)        // 16384
#define DMODEL  512
#define DINNER  1024
#define NSTATE  16
#define DTRANK  32
#define NLAYER  4

typedef __bf16 v16bf __attribute__((ext_vector_type(16)));
typedef float  v8f   __attribute__((ext_vector_type(8)));

__device__ __forceinline__ unsigned short f2bf(float f) {
    union { float f; unsigned u; } x; x.f = f;
    unsigned r = x.u + 0x7FFFu + ((x.u >> 16) & 1u);   // round-to-nearest-even
    return (unsigned short)(r >> 16);
}
__device__ __forceinline__ float bf2f(unsigned short h) {
    union { unsigned u; float f; } x; x.u = ((unsigned)h) << 16;
    return x.f;
}

// Issue two async B128 copies: 32 bytes global -> LDS (ASYNCcnt += 2).
// IOFFSET is added to both the LDS and global addresses (ISA 08_async_tensor).
__device__ __forceinline__ void async_copy32(unsigned lds_off,
                                             const unsigned short* gptr) {
    asm volatile(
        "global_load_async_to_lds_b128 %0, %1, off\n\t"
        "global_load_async_to_lds_b128 %0, %1, off offset:16"
        :: "v"(lds_off), "v"(gptr) : "memory");
}
__device__ __forceinline__ void wait_async0() {
    asm volatile("s_wait_asynccnt 0x0" ::: "memory");
}

// --------------------------- weight f32 -> bf16 ----------------------------
__global__ void cvt_bf16_kernel(const float* __restrict__ in,
                                unsigned short* __restrict__ out, int n) {
    int i = blockIdx.x * blockDim.x + threadIdx.x;
    if (i < n) out[i] = f2bf(in[i]);
}

// --------------------------- embedding (K=32) ------------------------------
__global__ void embed_kernel(const float* __restrict__ y,
                             const float* __restrict__ w,
                             const float* __restrict__ b,
                             float* __restrict__ h) {
    int idx = blockIdx.x * blockDim.x + threadIdx.x;
    if (idx >= M_TOK * DMODEL) return;
    int m = idx >> 9, d = idx & 511;
    const float* yr = y + (size_t)m * 32;
    const float* wr = w + (size_t)d * 32;
    float acc = b[d];
#pragma unroll
    for (int k = 0; k < 32; ++k) acc += yr[k] * wr[k];
    h[idx] = acc;
}

// --------------------------- layernorm -> bf16 -----------------------------
__global__ __launch_bounds__(256)
void ln_bf16_kernel(const float* __restrict__ in,
                    const float* __restrict__ w, const float* __restrict__ b,
                    unsigned short* __restrict__ out, int rows) {
    int lane = threadIdx.x & 31;
    int row  = blockIdx.x * 8 + (threadIdx.x >> 5);
    if (row >= rows) return;
    const float* x = in + (size_t)row * DMODEL;
    float v[16], s = 0.f, s2 = 0.f;
#pragma unroll
    for (int i = 0; i < 16; ++i) {
        float t = x[lane + i * 32]; v[i] = t; s += t; s2 += t * t;
    }
#pragma unroll
    for (int o = 16; o >= 1; o >>= 1) {
        s  += __shfl_xor(s,  o, 32);
        s2 += __shfl_xor(s2, o, 32);
    }
    float mean = s * (1.f / DMODEL);
    float var  = s2 * (1.f / DMODEL) - mean * mean;
    float rstd = rsqrtf(var + 1e-5f);
    unsigned short* yo = out + (size_t)row * DMODEL;
#pragma unroll
    for (int i = 0; i < 16; ++i) {
        int c = lane + i * 32;
        yo[c] = f2bf((v[i] - mean) * rstd * w[c] + b[c]);
    }
}

// --------------------------- WMMA GEMM -------------------------------------
// C[M,N] = act( A[M,K](bf16,row lda) * W[N,K]^T(bf16) + bias + residual )
// block = 256 threads (8 waves); tile 128(M) x 64(N); K step 32.
// A tiles double-buffered in LDS via async global->LDS DMA.
// act: 0=none, 1=softplus, 2=gelu(tanh)
__global__ __launch_bounds__(256)
void gemm_bf16_wmma(const unsigned short* __restrict__ A, int lda,
                    const unsigned short* __restrict__ W,
                    const float* __restrict__ bias,
                    const float* __restrict__ residual,
                    float* __restrict__ outF,
                    unsigned short* __restrict__ outB,
                    int M, int N, int K, int act) {
    __shared__ unsigned short As[2][128][40];  // padded rows (80B), 16B-aligned chunks

    const int tid   = threadIdx.x;
    const int lane  = tid & 31;
    const int wave  = tid >> 5;                 // 0..7
    const int mtile = blockIdx.y * 128;
    const int ntile = blockIdx.x * 64;
    const int lm    = lane & 15;
    const int kA    = (lane < 16) ? 0 : 8;      // A-frag chunk base (ISA layout)
    const int kB    = (lane < 16) ? 0 : 8;      // B-frag chunk base (matches A)

    v8f acc[4];
#pragma unroll
    for (int j = 0; j < 4; ++j)
#pragma unroll
        for (int r = 0; r < 8; ++r) acc[j][r] = 0.f;

    // each thread stages 16 bf16 (32B): row = tid/2, col half = (tid&1)*16
    const int arow = tid >> 1;                  // 0..127
    const int acol = (tid & 1) * 16;            // 0 or 16
    unsigned ldsA0 = (unsigned)(unsigned long long)&As[0][arow][acol];
    unsigned ldsA1 = (unsigned)(unsigned long long)&As[1][arow][acol];
    const unsigned short* ag = A + (size_t)(mtile + arow) * lda + acol;

    async_copy32(ldsA0, ag);                    // prefetch tile 0

    int buf = 0;
    for (int kk = 0; kk < K; kk += 32, buf ^= 1) {
        wait_async0();                          // my DMA for tile kk landed
        __syncthreads();                        // everyone's landed; prev buf free
        if (kk + 32 < K)                        // DMA next tile, overlaps WMMA
            async_copy32(buf ? ldsA0 : ldsA1, ag + kk + 32);

        // A fragment (16x32): lanes 0-15 K={0..7,16..23}, lanes 16-31 K={8..15,24..31}
        union { v16bf v; uint4 q[2]; } af;
        const unsigned short* asrc = &As[buf][wave * 16 + lm][0];
        af.q[0] = *((const uint4*)(asrc + kA));
        af.q[1] = *((const uint4*)(asrc + kA + 16));

#pragma unroll
        for (int j = 0; j < 4; ++j) {
            // B fragment (32x16): lane holds column N=lm, K={kB..kB+7, kB+16..kB+23}
            union { v16bf v; uint4 q[2]; } bfr;
            const unsigned short* wsrc =
                W + (size_t)(ntile + j * 16 + lm) * K + kk + kB;
            bfr.q[0] = *((const uint4*)wsrc);
            bfr.q[1] = *((const uint4*)(wsrc + 16));
            acc[j] = __builtin_amdgcn_wmma_f32_16x16x32_bf16(
                false, af.v, false, bfr.v, (short)0, acc[j], false, false);
        }
    }

    // epilogue: D layout -> VGPR r: M = r (lanes 0-15) / r+8 (lanes 16-31), N = lm
    const int rbase = mtile + wave * 16 + ((lane < 16) ? 0 : 8);
#pragma unroll
    for (int j = 0; j < 4; ++j) {
        int col  = ntile + j * 16 + lm;
        float bv = bias ? bias[col] : 0.f;
#pragma unroll
        for (int r = 0; r < 8; ++r) {
            int row    = rbase + r;
            size_t idx = (size_t)row * N + col;
            float v = acc[j][r] + bv;
            if (residual) v += residual[idx];
            if (act == 1) {                       // softplus
                v = (v > 20.f) ? v : log1pf(__expf(v));
            } else if (act == 2) {                // gelu (tanh approx)
                float u = v;
                v = 0.5f * u * (1.f + tanhf(0.7978845608f * (u + 0.044715f * u * u * u)));
            }
            if (outF) outF[idx] = v;
            if (outB) outB[idx] = f2bf(v);
        }
    }
}

// ------------------- causal depthwise conv (k=4) + SiLU --------------------
__global__ void conv_silu_kernel(const unsigned short* __restrict__ xz,
                                 const float* __restrict__ cw,
                                 const float* __restrict__ cb,
                                 unsigned short* __restrict__ xc) {
    int idx = blockIdx.x * blockDim.x + threadIdx.x;
    if (idx >= M_TOK * DINNER) return;
    int m = idx >> 10, d = idx & 1023;
    int l = m & (L_SEQ - 1);
    float acc = cb[d];
#pragma unroll
    for (int k = 0; k < 4; ++k) {
        int ll = l - 3 + k;
        if (ll >= 0)
            acc += cw[d * 4 + k] * bf2f(xz[(size_t)(m - 3 + k) * (2 * DINNER) + d]);
    }
    acc = acc / (1.f + __expf(-acc));             // silu
    xc[idx] = f2bf(acc);
}

// ----------------------------- selective scan ------------------------------
// Groups of 16 lanes share one channel d (lane%16 = state n). block 256 ->
// 16 channels; grid (DINNER/16, B). Sequential over L. Fused epilogue:
// y = (sum_n h*C + x*Dskip) * silu(z) -> bf16
__global__ __launch_bounds__(256)
void scan_kernel(const float* __restrict__ dbl,          // (M,64) f32
                 const unsigned short* __restrict__ dt,  // (M,DINNER) bf16
                 const unsigned short* __restrict__ xc,  // (M,DINNER) bf16
                 const unsigned short* __restrict__ xz,  // (M,2*DINNER) bf16 (z half)
                 const float* __restrict__ A_log,        // (DINNER,16)
                 const float* __restrict__ Dskip,        // (DINNER)
                 unsigned short* __restrict__ yout) {    // (M,DINNER) bf16
    int n = threadIdx.x & 15;
    int d = blockIdx.x * 16 + (threadIdx.x >> 4);
    int b = blockIdx.y;
    float Ad = -__expf(A_log[d * 16 + n]);
    float Dd = Dskip[d];
    float h  = 0.f;
    size_t base = (size_t)b * L_SEQ;
    for (int l = 0; l < L_SEQ; ++l) {
        size_t t   = base + l;
        float dtv  = bf2f(dt[t * DINNER + d]);
        float xv   = bf2f(xc[t * DINNER + d]);
        float Bn   = dbl[t * 64 + 32 + n];
        float Cn   = dbl[t * 64 + 48 + n];
        h = __expf(dtv * Ad) * h + (dtv * xv) * Bn;
        float yv = h * Cn;
        yv += __shfl_xor(yv, 8, 16);
        yv += __shfl_xor(yv, 4, 16);
        yv += __shfl_xor(yv, 2, 16);
        yv += __shfl_xor(yv, 1, 16);
        if (n == 0) {
            float zz = bf2f(xz[t * (2 * DINNER) + DINNER + d]);
            float s  = zz / (1.f + __expf(-zz));
            yout[t * DINNER + d] = f2bf((yv + xv * Dd) * s);
        }
    }
}

// ---------------------------------------------------------------------------
extern "C" void kernel_launch(void* const* d_in, const int* in_sizes, int n_in,
                              void* d_out, int out_size, void* d_ws, size_t ws_size,
                              hipStream_t stream) {
    (void)in_sizes; (void)n_in; (void)out_size; (void)ws_size;
    const float* y_in    = (const float*)d_in[0];
    const float* emb_w   = (const float*)d_in[1];
    const float* emb_b   = (const float*)d_in[2];
    const float* ln_w    = (const float*)d_in[3];
    const float* ln_b    = (const float*)d_in[4];
    const float* mix_w   = (const float*)d_in[5];
    const float* conv_w  = (const float*)d_in[6];
    const float* conv_b  = (const float*)d_in[7];
    const float* xproj_w = (const float*)d_in[8];
    const float* dtproj_w= (const float*)d_in[9];
    const float* dtproj_b= (const float*)d_in[10];
    const float* A_log   = (const float*)d_in[11];
    const float* Dskip   = (const float*)d_in[12];
    const float* out_w   = (const float*)d_in[13];
    const float* normf_w = (const float*)d_in[14];
    const float* normf_b = (const float*)d_in[15];
    const float* ro1_w   = (const float*)d_in[16];
    const float* ro1_b   = (const float*)d_in[17];
    const float* ro2_w   = (const float*)d_in[18];
    const float* ro2_b   = (const float*)d_in[19];

    char* ws = (char*)d_ws;
    size_t off = 0;
    auto sub = [&](size_t bytes) -> char* {
        char* p = ws + off;
        off += (bytes + 255) & ~(size_t)255;
        return p;
    };
    const size_t M = M_TOK;
    unsigned short* wMix = (unsigned short*)sub((size_t)NLAYER * 2048 * 512 * 2);
    unsigned short* wXp  = (unsigned short*)sub((size_t)NLAYER * 64 * 1024 * 2);
    unsigned short* wDt  = (unsigned short*)sub((size_t)NLAYER * 1024 * 32 * 2);
    unsigned short* wOut = (unsigned short*)sub((size_t)NLAYER * 512 * 1024 * 2);
    unsigned short* wR1  = (unsigned short*)sub((size_t)512 * 512 * 2);
    unsigned short* wR2  = (unsigned short*)sub((size_t)128 * 512 * 2);
    float*          hbuf = (float*)sub(M * DMODEL * 4);
    unsigned short* xin  = (unsigned short*)sub(M * DMODEL * 2);
    unsigned short* xz   = (unsigned short*)sub(M * 2 * DINNER * 2);
    unsigned short* xc   = (unsigned short*)sub(M * DINNER * 2);
    float*          dblF = (float*)sub(M * 64 * 4);
    unsigned short* dblB = (unsigned short*)sub(M * 64 * 2);
    unsigned short* dt   = (unsigned short*)sub(M * DINNER * 2);
    unsigned short* yb   = (unsigned short*)sub(M * DINNER * 2);

    auto cvt = [&](const float* s, unsigned short* d, int n) {
        cvt_bf16_kernel<<<(n + 255) / 256, 256, 0, stream>>>(s, d, n);
    };
    cvt(mix_w,    wMix, NLAYER * 2048 * 512);
    cvt(xproj_w,  wXp,  NLAYER * 64 * 1024);
    cvt(dtproj_w, wDt,  NLAYER * 1024 * 32);
    cvt(out_w,    wOut, NLAYER * 512 * 1024);
    cvt(ro1_w,    wR1,  512 * 512);
    cvt(ro2_w,    wR2,  128 * 512);

    embed_kernel<<<(M_TOK * DMODEL) / 256, 256, 0, stream>>>(y_in, emb_w, emb_b, hbuf);

    for (int li = 0; li < NLAYER; ++li) {
        ln_bf16_kernel<<<M_TOK / 8, 256, 0, stream>>>(
            hbuf, ln_w + li * DMODEL, ln_b + li * DMODEL, xin, M_TOK);
        // xz = xin @ mix_w^T          (M x 2048, K=512)
        gemm_bf16_wmma<<<dim3(2048 / 64, M_TOK / 128), 256, 0, stream>>>(
            xin, DMODEL, wMix + (size_t)li * 2048 * 512,
            nullptr, nullptr, nullptr, xz, M_TOK, 2048, 512, 0);
        conv_silu_kernel<<<(M_TOK * DINNER) / 256, 256, 0, stream>>>(
            xz, conv_w + li * DINNER * 4, conv_b + li * DINNER, xc);
        // dbl = xc @ xproj^T          (M x 64, K=1024)
        gemm_bf16_wmma<<<dim3(1, M_TOK / 128), 256, 0, stream>>>(
            xc, DINNER, wXp + (size_t)li * 64 * 1024,
            nullptr, nullptr, dblF, dblB, M_TOK, 64, 1024, 0);
        // dt = softplus(dt_r @ dtproj^T + b)   (M x 1024, K=32, lda=64)
        gemm_bf16_wmma<<<dim3(1024 / 64, M_TOK / 128), 256, 0, stream>>>(
            dblB, 64, wDt + (size_t)li * 1024 * 32,
            dtproj_b + li * DINNER, nullptr, nullptr, dt, M_TOK, 1024, 32, 1);
        scan_kernel<<<dim3(DINNER / 16, B_SZ), 256, 0, stream>>>(
            dblF, dt, xc, xz,
            A_log + (size_t)li * DINNER * NSTATE, Dskip + li * DINNER, yb);
        // h += y @ out_w^T            (M x 512, K=1024), residual fused
        gemm_bf16_wmma<<<dim3(512 / 64, M_TOK / 128), 256, 0, stream>>>(
            yb, DINNER, wOut + (size_t)li * 512 * 1024,
            nullptr, hbuf, hbuf, nullptr, M_TOK, 512, 1024, 0);
    }

    ln_bf16_kernel<<<M_TOK / 8, 256, 0, stream>>>(hbuf, normf_w, normf_b, xin, M_TOK);
    // g = gelu(h @ ro1^T + b)  -> bf16 (reuse yb)
    gemm_bf16_wmma<<<dim3(512 / 64, M_TOK / 128), 256, 0, stream>>>(
        xin, DMODEL, wR1, ro1_b, nullptr, nullptr, yb, M_TOK, 512, 512, 2);
    // out = g @ ro2^T + b  -> f32 d_out (M x 128)
    gemm_bf16_wmma<<<dim3(128 / 64, M_TOK / 128), 256, 0, stream>>>(
        yb, 512, wR2, ro2_b, nullptr, (float*)d_out, nullptr, M_TOK, 128, 512, 0);
}
